// MoE_share_gate_790273983070
// MI455X (gfx1250) — matrix operand order repaired and checked by reading, not compiled, not run on hardware
//
#include <hip/hip_runtime.h>
#include <hip/hip_bf16.h>

typedef __attribute__((ext_vector_type(16))) _Float16 v16h;
typedef __attribute__((ext_vector_type(8)))  _Float16 v8h;
typedef __attribute__((ext_vector_type(8)))  float    v8f;
typedef __attribute__((ext_vector_type(4)))  unsigned int u32x4;
typedef __attribute__((ext_vector_type(8)))  int      i32x8;
typedef __attribute__((ext_vector_type(4)))  int      i32x4;

static constexpr int BROWS = 65536;
static constexpr int DDIM  = 128;
static constexpr int HDIM  = 256;
static constexpr int ODIM  = 128;
static constexpr int NEXP  = 8;
// LDS row pitches (halfs): +8 halfs (16B) so consecutive rows rotate LDS banks
static constexpr int W1PITCH = DDIM + 8;   // 136 halfs = 272 B
static constexpr int W2PITCH = HDIM + 8;   // 264 halfs = 528 B
static constexpr int HPAD    = HDIM + 8;   // 264

#if defined(__has_builtin)
#if __has_builtin(__builtin_amdgcn_tensor_load_to_lds)
#define HAVE_TDM 1
#else
#define HAVE_TDM 0
#endif
#else
#define HAVE_TDM 0
#endif

__device__ __forceinline__ v16h cat8(v8h a, v8h b) {
  return __builtin_shufflevector(a, b, 0,1,2,3,4,5,6,7,8,9,10,11,12,13,14,15);
}

#if HAVE_TDM
#if __has_builtin(__builtin_amdgcn_s_wait_tensorcnt)
#define WAIT_TENSORCNT(n) __builtin_amdgcn_s_wait_tensorcnt((short)(n))
#else
#define WAIT_TENSORCNT(n) asm volatile("s_wait_tensorcnt %0" ::"i"(n) : "memory")
#endif

// 1D-row TDM load: `rows` rows of `row_units` 8-byte units each, contiguous in
// global memory, written to LDS with padding inserted after every
// 2^(pad_interval_code+1) DWORDs (i.e. after each row): pad_amount_code+1 DWORDs.
__device__ __forceinline__ void tdm_load_rows(unsigned lds_addr, const void* gptr,
                                              unsigned rows, unsigned row_units,
                                              unsigned pad_interval_code,
                                              unsigned pad_amount_code) {
  unsigned long long ga = (unsigned long long)(size_t)gptr;
  u32x4 g0;
  g0[0] = 1u;                                            // count=1, user descriptor
  g0[1] = lds_addr;                                      // lds_addr -> bits[63:32]
  g0[2] = (unsigned)ga;                                  // global_addr[31:0]
  g0[3] = (unsigned)((ga >> 32) & 0x1FFFFFFu) | (2u << 30); // addr hi | type=2
  i32x8 g1;
  // data_size=3 (8B) | pad_enable | pad_interval | pad_amount
  g1[0] = (int)((3u << 16) | (1u << 20) | (pad_interval_code << 22) |
                (pad_amount_code << 25));
  g1[1] = (int)((row_units & 0xFFFFu) << 16);            // tensor_dim0[15:0]
  g1[2] = (int)((row_units >> 16) | ((rows & 0xFFFFu) << 16)); // dim0 hi | dim1 lo
  g1[3] = (int)((row_units & 0xFFFFu) << 16);            // tile_dim0
  g1[4] = (int)rows;                                     // tile_dim1 (tile_dim2=0)
  g1[5] = (int)row_units;                                // tensor_dim0_stride lo32
  g1[6] = 0;
  g1[7] = 0;
  i32x4 z4 = {0, 0, 0, 0};
#if __clang_major__ >= 23
  i32x8 z8 = {0, 0, 0, 0, 0, 0, 0, 0};
  __builtin_amdgcn_tensor_load_to_lds(g0, g1, z4, z4, z8, 0);
#else
  __builtin_amdgcn_tensor_load_to_lds(g0, g1, z4, z4, 0);
#endif
}
#endif  // HAVE_TDM

// ---------------- gating: logits -> top2 -> softmax -> dense gates[B][8] ----
__global__ __launch_bounds__(256) void gate_kernel(const float* __restrict__ x,
                                                   const float* __restrict__ wg,
                                                   float* __restrict__ gates) {
  int b = blockIdx.x * blockDim.x + threadIdx.x;
  if (b >= BROWS) return;
  float logit[NEXP];
  #pragma unroll
  for (int e = 0; e < NEXP; ++e) logit[e] = 0.f;
  const float* xr = x + (size_t)b * DDIM;
  for (int d = 0; d < DDIM; ++d) {
    float xv = xr[d];
    #pragma unroll
    for (int e = 0; e < NEXP; ++e) logit[e] += xv * wg[d * NEXP + e];
  }
  int i0 = 0; float v0 = logit[0];
  #pragma unroll
  for (int e = 1; e < NEXP; ++e) if (logit[e] > v0) { v0 = logit[e]; i0 = e; }
  int i1 = -1; float v1 = -3.4e38f;
  #pragma unroll
  for (int e = 0; e < NEXP; ++e) if (e != i0 && logit[e] > v1) { v1 = logit[e]; i1 = e; }
  float e1 = __expf(v1 - v0);          // v0 >= v1
  float inv = 1.f / (1.f + e1);
  float g[NEXP];
  #pragma unroll
  for (int e = 0; e < NEXP; ++e) g[e] = 0.f;
  g[i0] = inv;
  g[i1] = e1 * inv;
  #pragma unroll
  for (int e = 0; e < NEXP; ++e) gates[(size_t)b * NEXP + e] = g[e];
}

// ---------------- prep: f32 -> f16 copies / transposed weights ---------------
__global__ __launch_bounds__(256) void conv_f16_kernel(const float* __restrict__ in,
                                                       _Float16* __restrict__ out, int n) {
  int i = blockIdx.x * blockDim.x + threadIdx.x;
  if (i < n) out[i] = (_Float16)in[i];
}

// w1[e][d][h] -> w1t[e][h][d]
__global__ __launch_bounds__(256) void transpose_w1_kernel(const float* __restrict__ w1,
                                                           _Float16* __restrict__ w1t) {
  int i = blockIdx.x * blockDim.x + threadIdx.x;
  if (i >= NEXP * DDIM * HDIM) return;
  int e = i / (DDIM * HDIM);
  int r = i % (DDIM * HDIM);
  int h = r / DDIM;
  int d = r % DDIM;
  w1t[i] = (_Float16)w1[((size_t)e * DDIM + d) * HDIM + h];
}

// w2[e][h][o] -> w2t[e][o][h]
__global__ __launch_bounds__(256) void transpose_w2_kernel(const float* __restrict__ w2,
                                                           _Float16* __restrict__ w2t) {
  int i = blockIdx.x * blockDim.x + threadIdx.x;
  if (i >= NEXP * HDIM * ODIM) return;
  int e = i / (HDIM * ODIM);
  int r = i % (HDIM * ODIM);
  int o = r / HDIM;
  int h = r % HDIM;
  w2t[i] = (_Float16)w2[((size_t)e * HDIM + h) * ODIM + o];
}

// ------- fused MoE: 16-row tiles, dense over experts, pipelined TDM staging --
__global__ __launch_bounds__(256) void moe_wmma_kernel(
    const _Float16* __restrict__ xh,
    const _Float16* __restrict__ w1t,
    const float*    __restrict__ b1,
    const _Float16* __restrict__ w2t,
    const float*    __restrict__ b2,
    const float*    __restrict__ gates,
    float*          __restrict__ out) {
  __shared__ alignas(16) _Float16 w1s[HDIM * W1PITCH];   // 68 KB staged w1t[e]
  __shared__ alignas(16) _Float16 w2s[ODIM * W2PITCH];   // 66 KB staged w2t[e]
  __shared__ alignas(16) _Float16 hbuf[8][16][HPAD];     // 66 KB h tiles

  const int wave = threadIdx.x >> 5;
  const int lane = threadIdx.x & 31;
  const int hi   = lane >> 4;     // half-wave select
  const int l15  = lane & 15;
  const int b0   = blockIdx.x * 128 + wave * 16;

  // Preload x A-fragments (16 rows x K=128 as 4 chunks of 32); reused by all experts.
  // A-layout (16-bit, 16x32): lanes 0-15 row M=lane hold K {0..7,16..23};
  // lanes 16-31 same rows hold K {8..15,24..31}.
  v16h ax[4];
  {
    const _Float16* xrow = xh + (size_t)(b0 + l15) * DDIM + hi * 8;
    #pragma unroll
    for (int kt = 0; kt < 4; ++kt) {
      const _Float16* p = xrow + kt * 32;
      ax[kt] = cat8(*(const v8h*)p, *(const v8h*)(p + 16));
    }
  }

  float comb[8][8];
  #pragma unroll
  for (int ot = 0; ot < 8; ++ot)
    #pragma unroll
    for (int g = 0; g < 8; ++g) comb[ot][g] = 0.f;

#if HAVE_TDM
  // Pipeline prologue: start streaming expert 0's first-layer weights.
  if (threadIdx.x < 32) {
    tdm_load_rows((unsigned)(size_t)&w1s[0], w1t, HDIM, DDIM / 4, 5u, 3u);
  }
#endif

  for (int e = 0; e < NEXP; ++e) {
    if (e + 1 < NEXP) {
      __builtin_prefetch(w1t + (size_t)(e + 1) * HDIM * DDIM, 0, 1);
      __builtin_prefetch(w2t + (size_t)(e + 1) * ODIM * HDIM, 0, 1);
    }
#if HAVE_TDM
    if (threadIdx.x < 32) {
      // Stream w2s(e) behind layer-1 compute; TDM completes in order per wave,
      // so TENSORcnt<=1 guarantees the older w1s(e) load has landed.
      tdm_load_rows((unsigned)(size_t)&w2s[0],
                    w2t + (size_t)e * ODIM * HDIM, ODIM, HDIM / 4, 6u, 3u);
      WAIT_TENSORCNT(1);
    }
#else
    {
      const _Float16* s1 = w1t + (size_t)e * HDIM * DDIM;
      for (int i = threadIdx.x; i < HDIM * (DDIM / 8); i += 256) {
        int r = i / (DDIM / 8), c = i % (DDIM / 8);
        *(v8h*)(w1s + r * W1PITCH + c * 8) = *(const v8h*)(s1 + r * DDIM + c * 8);
      }
      const _Float16* s2 = w2t + (size_t)e * ODIM * HDIM;
      for (int i = threadIdx.x; i < ODIM * (HDIM / 8); i += 256) {
        int r = i / (HDIM / 8), c = i % (HDIM / 8);
        *(v8h*)(w2s + r * W2PITCH + c * 8) = *(const v8h*)(s2 + r * HDIM + c * 8);
      }
    }
#endif
    __syncthreads();   // (A) publish w1s(e)

    // ---- layer 1: h = relu(x @ w1[e] + b1[e]); f16 tile -> LDS -------------
    for (int nt = 0; nt < HDIM / 16; ++nt) {
      v8f acc = {};
      // B-layout (16-bit, 32x16): lane holds column N=l15, K = hi*8 + {0..7,16..23}
      const _Float16* wrow = w1s + (nt * 16 + l15) * W1PITCH + hi * 8;
      #pragma unroll
      for (int kt = 0; kt < 4; ++kt) {
        const _Float16* p = wrow + kt * 32;
        v16h bf = cat8(*(const v8h*)p, *(const v8h*)(p + 16));
        acc = __builtin_amdgcn_wmma_f32_16x16x32_f16(false, ax[kt], false, bf,
                                                     (short)0, acc, false, false);
      }
      // C-layout: col N = l15, rows M = g + 8*hi
      const int col = nt * 16 + l15;
      const float bias = b1[e * HDIM + col];
      #pragma unroll
      for (int g = 0; g < 8; ++g) {
        float v = acc[g] + bias;
        v = v > 0.f ? v : 0.f;
        hbuf[wave][g + 8 * hi][col] = (_Float16)v;
      }
    }
    __syncthreads();   // (B) hbuf ready; w1s free for overwrite

#if HAVE_TDM
    if (threadIdx.x < 32) {
      if (e + 1 < NEXP) {
        // Stream w1s(e+1) behind layer-2 compute; TENSORcnt<=1 guarantees the
        // older w2s(e) load has landed.
        tdm_load_rows((unsigned)(size_t)&w1s[0],
                      w1t + (size_t)(e + 1) * HDIM * DDIM, HDIM, DDIM / 4, 5u, 3u);
        WAIT_TENSORCNT(1);
      } else {
        WAIT_TENSORCNT(0);
      }
    }
#endif
    __syncthreads();   // (C) publish w2s(e)

    float gate_r[8];
    #pragma unroll
    for (int g = 0; g < 8; ++g)
      gate_r[g] = gates[(size_t)(b0 + g + 8 * hi) * NEXP + e];

    // ---- layer 2: o = h @ w2[e] + b2[e]; comb += gate * exp(o) -------------
    #pragma unroll
    for (int ot = 0; ot < ODIM / 16; ++ot) {
      v8f acc = {};
      const _Float16* w2row = w2s + (ot * 16 + l15) * W2PITCH + hi * 8;
      const _Float16* hrow  = &hbuf[wave][l15][hi * 8];
      #pragma unroll
      for (int kt = 0; kt < 8; ++kt) {
        const _Float16* pa = hrow + kt * 32;
        v16h af = cat8(*(const v8h*)pa, *(const v8h*)(pa + 16));
        const _Float16* pb = w2row + kt * 32;
        v16h bf = cat8(*(const v8h*)pb, *(const v8h*)(pb + 16));
        acc = __builtin_amdgcn_wmma_f32_16x16x32_f16(false, af, false, bf,
                                                     (short)0, acc, false, false);
      }
      const int ocol = ot * 16 + l15;
      const float bias2 = b2[e * ODIM + ocol];
      #pragma unroll
      for (int g = 0; g < 8; ++g)
        comb[ot][g] += gate_r[g] * __expf(acc[g] + bias2);
    }
    __syncthreads();   // (D) w2s reads done before next iteration's TDM write
  }

  // ---- combine: log(comb == 0 ? EPS : comb) ----------------------------------------
  #pragma unroll
  for (int ot = 0; ot < 8; ++ot) {
    const int ocol = ot * 16 + l15;
    #pragma unroll
    for (int g = 0; g < 8; ++g) {
      float v = comb[ot][g];
      v = (v == 0.f) ? 2.2204460492503131e-16f : v;
      out[(size_t)(b0 + g + 8 * hi) * ODIM + ocol] = __logf(v);
    }
  }
}

extern "C" void kernel_launch(void* const* d_in, const int* in_sizes, int n_in,
                              void* d_out, int out_size, void* d_ws, size_t ws_size,
                              hipStream_t stream) {
  const float* x  = (const float*)d_in[0];
  const float* wg = (const float*)d_in[1];
  const float* w1 = (const float*)d_in[2];
  const float* b1 = (const float*)d_in[3];
  const float* w2 = (const float*)d_in[4];
  const float* b2 = (const float*)d_in[5];
  float* out = (float*)d_out;

  char* ws = (char*)d_ws;
  float*    gates = (float*)ws;                                   // 2 MB
  _Float16* xh    = (_Float16*)(ws + (size_t)2  * 1024 * 1024);   // 16 MB
  _Float16* w1t   = (_Float16*)(ws + (size_t)18 * 1024 * 1024);   // 512 KB
  _Float16* w2t   = (_Float16*)(ws + (size_t)18 * 1024 * 1024 + 512 * 1024);

  gate_kernel<<<BROWS / 256, 256, 0, stream>>>(x, wg, gates);
  conv_f16_kernel<<<(BROWS * DDIM) / 256, 256, 0, stream>>>(x, xh, BROWS * DDIM);
  transpose_w1_kernel<<<(NEXP * DDIM * HDIM) / 256, 256, 0, stream>>>(w1, w1t);
  transpose_w2_kernel<<<(NEXP * HDIM * ODIM) / 256, 256, 0, stream>>>(w2, w2t);
  moe_wmma_kernel<<<BROWS / 128, 256, 0, stream>>>(xh, w1t, b1, w2t, b2, gates, out);
}
